// SVJNet_41601053229897
// MI455X (gfx1250) — compile-verified, matrix operand order checked by reading, not run on hardware
//
#include <hip/hip_runtime.h>

#define NGRAPH 512
#define P      128
#define HID    64
#define KNN    24
#define NT     256

typedef __attribute__((ext_vector_type(16))) _Float16 v16h;
typedef __attribute__((ext_vector_type(8)))  float    v8f;

__device__ __forceinline__ float elu_f(float x) {
  return x > 0.0f ? x : (__expf(x) - 1.0f);
}

// ---------------------------------------------------------------------------
// Fragment-swizzled LDS storage.
// WMMA 16x16x32 f16 operand layout (ISA 7.12.2): lane L holds row/col (L&15);
// element e -> K = ((L&16)>>1) + (e&7) + ((e&8)<<1), per 32-wide K-half.
// Each (tile, khalf, lane) 16-half vector is stored contiguously so a
// fragment load is ONE aligned 32-byte LDS read (ds_load_b128 x2).
//   H  slots: (tile[0..7]*2 + khalf)*32 + lane      -> 512 slots * 32B = 16KB
//   W  slots: (ntile[0..3]*2 + khalf)*32 + lane     -> 256 slots * 32B =  8KB
// ---------------------------------------------------------------------------

__device__ __forceinline__ v16h ldfrag(const _Float16* base, int tile, int khalf, int lane) {
  return *(const v16h*)(base + (((tile << 1) + khalf) * 32 + lane) * 16);
}

// Build swizzled f16 fragments from row-major f32 src[128][64]. 512 slots,
// each thread builds 2. Gather = two contiguous 8-float runs (vectorizable).
__device__ __forceinline__ void build_h_frags(const float* src, _Float16* dst, int tid) {
#pragma unroll
  for (int t = 0; t < 2; ++t) {
    const int s     = tid + t * NT;            // 0..511
    const int tile  = s >> 6;
    const int khalf = (s >> 5) & 1;
    const int lslot = s & 31;
    const int row   = tile * 16 + (lslot & 15);
    const int kbase = khalf * 32 + ((lslot & 16) >> 1);
    const float* p  = src + row * HID + kbase;
    v16h f;
#pragma unroll
    for (int e = 0; e < 16; ++e)
      f[e] = (_Float16)p[(e & 7) + ((e & 8) << 1)];
    *(v16h*)(dst + s * 16) = f;
  }
}

// Swizzle one 64x64 f32 weight matrix (row-major, global) into fragments.
__device__ __forceinline__ void build_w_frags(const float* w, _Float16* dst, int tid) {
  if (tid < 256) {
    const int s     = tid;                     // 0..255
    const int n     = ((s >> 6) << 4) + (s & 15);
    const int kbase = (((s >> 5) & 1) << 5) + ((s & 16) >> 1);
    v16h f;
#pragma unroll
    for (int e = 0; e < 16; ++e) {
      const int k = kbase + (e & 7) + ((e & 8) << 1);
      f[e] = (_Float16)w[k * HID + n];
    }
    *(v16h*)(dst + s * 16) = f;
  }
}

// Swizzle both edge-conv matrices at once, sharing global loads:
// W1 = wc[0:64,:] - wc[64:128,:] ; W2 = wc[64:128,:]
__device__ __forceinline__ void build_wc_frags(const float* wc, _Float16* w0, _Float16* w1, int tid) {
  if (tid < 256) {
    const int s     = tid;
    const int n     = ((s >> 6) << 4) + (s & 15);
    const int kbase = (((s >> 5) & 1) << 5) + ((s & 16) >> 1);
    v16h f1, f2;
#pragma unroll
    for (int e = 0; e < 16; ++e) {
      const int k = kbase + (e & 7) + ((e & 8) << 1);
      const float a = wc[k * HID + n];
      const float b = wc[(HID + k) * HID + n];
      f1[e] = (_Float16)(a - b);
      f2[e] = (_Float16)b;
    }
    *(v16h*)(w0 + s * 16) = f1;
    *(v16h*)(w1 + s * 16) = f2;
  }
}

// C[128][64] = H[128][64] @ W[64][64] (+bias, optional ELU). Wave w owns row
// tile w; 4 col tiles x 2 K-steps = 8 v_wmma per wave.
__device__ __forceinline__ void mm128(const _Float16* hhs, const _Float16* ws,
                                      const float* bias, bool do_elu,
                                      float* C, int wave, int lane) {
  const v16h a0 = ldfrag(hhs, wave, 0, lane);
  const v16h a1 = ldfrag(hhs, wave, 1, lane);
  const int rbase = wave * 16 + ((lane & 16) >> 1);
  const int cl    = lane & 15;
#pragma unroll
  for (int s = 0; s < 4; ++s) {
    v8f acc = {};
    v16h b0 = ldfrag(ws, s, 0, lane);
    acc = __builtin_amdgcn_wmma_f32_16x16x32_f16(false, a0, false, b0, (short)0, acc, false, false);
    v16h b1 = ldfrag(ws, s, 1, lane);
    acc = __builtin_amdgcn_wmma_f32_16x16x32_f16(false, a1, false, b1, (short)0, acc, false, false);
    const int   c  = s * 16 + cl;
    const float bv = bias ? bias[c] : 0.0f;
#pragma unroll
    for (int j = 0; j < 8; ++j) {
      float v = acc[j] + bv;
      if (do_elu) v = elu_f(v);
      C[(rbase + j) * HID + c] = v;
    }
  }
}

// G[128][128] = H @ H^T. B tile = rows of H -> same fragment layout as A.
__device__ __forceinline__ void gram128(const _Float16* hhs, float* G, int wave, int lane) {
  const v16h a0 = ldfrag(hhs, wave, 0, lane);
  const v16h a1 = ldfrag(hhs, wave, 1, lane);
  const int rbase = wave * 16 + ((lane & 16) >> 1);
  const int cl    = lane & 15;
#pragma unroll
  for (int s = 0; s < 8; ++s) {
    v8f acc = {};
    v16h b0 = ldfrag(hhs, s, 0, lane);
    acc = __builtin_amdgcn_wmma_f32_16x16x32_f16(false, a0, false, b0, (short)0, acc, false, false);
    v16h b1 = ldfrag(hhs, s, 1, lane);
    acc = __builtin_amdgcn_wmma_f32_16x16x32_f16(false, a1, false, b1, (short)0, acc, false, false);
#pragma unroll
    for (int j = 0; j < 8; ++j)
      G[(rbase + j) * P + s * 16 + cl] = acc[j];
  }
}

struct SMem {
  float h[P][HID];                       // 32 KB current features f32
  float g[P][P];                         // 64 KB Gram/dist, reused as U|V
  __align__(32) _Float16 hhs[512 * 16];  // 16 KB swizzled H fragments
  __align__(32) _Float16 w0[256 * 16];   //  8 KB swizzled weight frags
  __align__(32) _Float16 w1[256 * 16];   //  8 KB
  float          sq[P];
  unsigned short knn[P][KNN];            //  6 KB
  float          pooled[HID];
  float          tmp[HID];
};                                       // ~138 KB

__global__ __launch_bounds__(NT) void svjnet_kernel(
    const float* __restrict__ x_pf, const long long* __restrict__ batch,
    const float* __restrict__ we1, const float* __restrict__ be1,
    const float* __restrict__ we2, const float* __restrict__ be2,
    const float* __restrict__ wc1, const float* __restrict__ bc1,
    const float* __restrict__ wc2, const float* __restrict__ bc2,
    const float* __restrict__ wc3, const float* __restrict__ bc3,
    const float* __restrict__ wo1, const float* __restrict__ bo1,
    const float* __restrict__ wo2, const float* __restrict__ bo2,
    const float* __restrict__ wo3, const float* __restrict__ bo3,
    const float* __restrict__ wo4, const float* __restrict__ bo4,
    const float* __restrict__ wo5, const float* __restrict__ bo5,
    float* __restrict__ out) {
  __shared__ SMem sm;
  const int tid  = threadIdx.x;
  const int lane = tid & 31;
  const int wave = tid >> 5;
  const int g    = blockIdx.x;
  const float* xg = x_pf + (size_t)g * P * 4;

  float* hF = &sm.h[0][0];
  float* gF = &sm.g[0][0];

  // Warm L2/WGP$ for the per-layer weight swizzles (gfx1250 global_prefetch_b8).
  __builtin_prefetch((const char*)wc1 + tid * 128, 0, 1);
  __builtin_prefetch((const char*)wc2 + tid * 128, 0, 1);
  __builtin_prefetch((const char*)wc3 + tid * 128, 0, 1);

  // ---- encoder layer 1: elu(x @ we1 + be1), K=4, plain VALU ----
#pragma unroll 1
  for (int idx = tid; idx < P * HID; idx += NT) {
    const int p = idx >> 6, c = idx & 63;
    float acc = be1[c];
#pragma unroll
    for (int k = 0; k < 4; ++k) acc = fmaf(xg[p * 4 + k], we1[k * HID + c], acc);
    hF[idx] = elu_f(acc);
  }
  build_w_frags(we2, sm.w0, tid);
  __syncthreads();
  build_h_frags(hF, sm.hhs, tid);
  __syncthreads();

  // ---- encoder layer 2: elu(h1 @ we2 + be2) via WMMA ----
  mm128(sm.hhs, sm.w0, be2, true, hF, wave, lane);
  __syncthreads();
  build_h_frags(hF, sm.hhs, tid);
  __syncthreads();

  // ---- 3x DynamicEdgeConv ----
#pragma unroll 1
  for (int layer = 0; layer < 3; ++layer) {
    const float* wc = (layer == 0) ? wc1 : (layer == 1) ? wc2 : wc3;
    const float* bc = (layer == 0) ? bc1 : (layer == 1) ? bc2 : bc3;

    build_wc_frags(wc, sm.w0, sm.w1, tid);

    // Gram matrix G = H H^T via WMMA (f16 data, f32 accum)
    gram128(sm.hhs, gF, wave, lane);
    __syncthreads();
    if (tid < P) sm.sq[tid] = sm.g[tid][tid];
    __syncthreads();

    // top-K smallest d_ij (incl. self, d_ii == 0). One pass converts this
    // thread's G row to distances in-place; 24 scan rounds are then bare
    // load+min, with picked entries poisoned to +inf. Unrolling clamped:
    // 8 waves already hide LDS latency, full unroll just bloats I-cache.
    if (tid < P) {
      const float si = sm.sq[tid];
      float* grow = &sm.g[tid][0];
#pragma unroll 4
      for (int j = 0; j < P; ++j)
        grow[j] = si + sm.sq[j] - 2.0f * grow[j];
#pragma unroll 1
      for (int k = 0; k < KNN; ++k) {
        float best = 3.4e38f; int bj = 0;
#pragma unroll 4
        for (int j = 0; j < P; ++j) {
          const float d = grow[j];
          if (d < best) { best = d; bj = j; }
        }
        grow[bj] = 3.5e38f;                // excluded from later rounds
        sm.knn[tid][k] = (unsigned short)bj;
      }
    }
    __syncthreads();

    // U = H @ W1 ; V = H @ W2  (reuse Gram buffer: 2 x 128*64 f32)
    float* U = gF;
    float* V = gF + P * HID;
    mm128(sm.hhs, sm.w0, nullptr, false, U, wave, lane);
    mm128(sm.hhs, sm.w1, nullptr, false, V, wave, lane);
    __syncthreads();

    // ELU is strictly increasing, so max_j elu(u+v_j) = elu(u + max_j v_j):
    // inner loop is pure v_max_f32, one ELU per output.
#pragma unroll 1
    for (int idx = tid; idx < P * HID; idx += NT) {
      const int i = idx >> 6, c = idx & 63;
      float m = -3.4e38f;
#pragma unroll 4
      for (int k = 0; k < KNN; ++k) {
        const int j = sm.knn[i][k];
        m = fmaxf(m, V[j * HID + c]);
      }
      hF[idx] = elu_f(U[i * HID + c] + bc[c] + m);
    }
    __syncthreads();
    build_h_frags(hF, sm.hhs, tid);
    __syncthreads();
  }

  // ---- sum-pool over nodes ----
  if (tid < HID) {
    float s = 0.0f;
#pragma unroll 4
    for (int i = 0; i < P; ++i) s += sm.h[i][tid];
    sm.pooled[tid] = s;
  }
  __syncthreads();

  // ---- MLP head (tiny, VALU) ----
  if (tid < 64) {
    float acc = bo1[tid];
#pragma unroll 4
    for (int k = 0; k < 64; ++k) acc = fmaf(sm.pooled[k], wo1[k * 64 + tid], acc);
    sm.tmp[tid] = elu_f(acc);
  }
  __syncthreads();
  if (tid < 32) {
    float acc = bo2[tid];
#pragma unroll 4
    for (int k = 0; k < 64; ++k) acc = fmaf(sm.tmp[k], wo2[k * 32 + tid], acc);
    sm.pooled[tid] = elu_f(acc);
  }
  __syncthreads();
  if (tid < 32) {
    float acc = bo3[tid];
#pragma unroll 4
    for (int k = 0; k < 32; ++k) acc = fmaf(sm.pooled[k], wo3[k * 32 + tid], acc);
    sm.tmp[tid] = elu_f(acc);
  }
  __syncthreads();
  if (tid < 8) {
    float acc = bo4[tid];
#pragma unroll 4
    for (int k = 0; k < 32; ++k) acc = fmaf(sm.tmp[k], wo4[k * 8 + tid], acc);
    sm.pooled[tid] = elu_f(acc);
  }
  __syncthreads();
  if (tid == 0) {
    float acc = bo5[0];
#pragma unroll
    for (int k = 0; k < 8; ++k) acc = fmaf(sm.pooled[k], wo5[k], acc);
    out[g] = acc;
  }

  // ---- tuple output #2: batch_pf passthrough (written as float) ----
#pragma unroll 1
  for (int idx = tid; idx < P; idx += NT)
    out[NGRAPH + (size_t)g * P + idx] = (float)batch[(size_t)g * P + idx];
}

extern "C" void kernel_launch(void* const* d_in, const int* in_sizes, int n_in,
                              void* d_out, int out_size, void* d_ws, size_t ws_size,
                              hipStream_t stream) {
  (void)in_sizes; (void)n_in; (void)out_size; (void)d_ws; (void)ws_size;
  const float*     x_pf  = (const float*)d_in[0];
  const long long* batch = (const long long*)d_in[1];
  const float* we1 = (const float*)d_in[2];
  const float* be1 = (const float*)d_in[3];
  const float* we2 = (const float*)d_in[4];
  const float* be2 = (const float*)d_in[5];
  const float* wc1 = (const float*)d_in[6];
  const float* bc1 = (const float*)d_in[7];
  const float* wc2 = (const float*)d_in[8];
  const float* bc2 = (const float*)d_in[9];
  const float* wc3 = (const float*)d_in[10];
  const float* bc3 = (const float*)d_in[11];
  const float* wo1 = (const float*)d_in[12];
  const float* bo1 = (const float*)d_in[13];
  const float* wo2 = (const float*)d_in[14];
  const float* bo2 = (const float*)d_in[15];
  const float* wo3 = (const float*)d_in[16];
  const float* bo3 = (const float*)d_in[17];
  const float* wo4 = (const float*)d_in[18];
  const float* bo4 = (const float*)d_in[19];
  const float* wo5 = (const float*)d_in[20];
  const float* bo5 = (const float*)d_in[21];

  svjnet_kernel<<<NGRAPH, NT, 0, stream>>>(
      x_pf, batch, we1, be1, we2, be2, wc1, bc1, wc2, bc2, wc3, bc3,
      wo1, bo1, wo2, bo2, wo3, bo3, wo4, bo4, wo5, bo5, (float*)d_out);
}
